// TF2Model_9234179686965
// MI455X (gfx1250) — matrix operand order repaired and checked by reading, not compiled, not run on hardware
//
#include <hip/hip_runtime.h>
#include <hip/hip_bf16.h>
#include <stdint.h>

typedef __attribute__((ext_vector_type(16))) __bf16 v16bf;
typedef __attribute__((ext_vector_type(8)))  __bf16 v8bf;
typedef __attribute__((ext_vector_type(8)))  float  v8f;

#define UNITS 100
#define FEAT  256
#define BATCH 128
#define TIME  1024
#define NG    400      // 4*UNITS
#define NT    25       // N tiles of 16 (25*16 == 400)
#define HPAD  136      // padded h row stride (bf16) -> 272B stride, 17-DW bank walk
#define ZBLK  (16 * NG)          // floats per (wg, t) slice of Z1pre
#define ZBLKB (ZBLK * 4)         // 25600 bytes

// ---- fast transcendentals: CDNA5 has V_TANH_F32 in hardware ---------------
#if __has_builtin(__builtin_amdgcn_tanhf)
__device__ __forceinline__ float fast_tanh(float x) { return __builtin_amdgcn_tanhf(x); }
#elif __has_builtin(__builtin_amdgcn_tanh_f32)
__device__ __forceinline__ float fast_tanh(float x) { return __builtin_amdgcn_tanh_f32(x); }
#else
__device__ __forceinline__ float fast_tanh(float x) { return tanhf(x); }
#endif
// sigmoid(x) == 0.5*tanh(x/2) + 0.5 (exact identity) -> one TRANS op
__device__ __forceinline__ float fast_sigmoid(float x) {
  return __builtin_fmaf(0.5f, fast_tanh(0.5f * x), 0.5f);
}

// ---------------------------------------------------------------------------
// Kernel 0: convert weights fp32 -> bf16 (tiny, one shot)
// ---------------------------------------------------------------------------
__global__ void __launch_bounds__(256)
convert_weights(const float* __restrict__ W1, const float* __restrict__ U1,
                const float* __restrict__ W2, const float* __restrict__ U2,
                __bf16* __restrict__ W1b, __bf16* __restrict__ U1b,
                __bf16* __restrict__ W2b, __bf16* __restrict__ U2b) {
  int i = blockIdx.x * 256 + threadIdx.x;
  if (i < FEAT * NG) W1b[i] = (__bf16)W1[i];
  if (i < UNITS * NG) {
    U1b[i] = (__bf16)U1[i];
    W2b[i] = (__bf16)W2[i];
    U2b[i] = (__bf16)U2[i];
  }
}

// ---------------------------------------------------------------------------
// Kernel 1: Z1pre = x @ W1 + b1, stored scan-blocked: [wg][t][16 batch][400]
// M = B*T = 131072 rows, N = 400, K = 256. One 16x16 tile per wave.
// ---------------------------------------------------------------------------
__global__ void __launch_bounds__(256)
input_proj(const float* __restrict__ x, const __bf16* __restrict__ W1b,
           const float* __restrict__ b1, float* __restrict__ Z1) {
  const int lane = threadIdx.x & 31;
  const int wave = threadIdx.x >> 5;
  const int gid  = blockIdx.x * 8 + wave;          // global wave-tile id
  const int tN   = gid % NT;
  const int tM   = gid / NT;
  const int m16  = lane & 15;
  const int hi   = lane >> 4;
  const int n0   = tN * 16;
  const int row  = tM * 16 + m16;                  // flat (b,t) row for A

  v8f c;
  {
    float bias = b1[n0 + m16];
#pragma unroll
    for (int r = 0; r < 8; ++r) c[r] = bias;      // fold b1 into accumulator
  }

  const float* xr = x + (size_t)row * FEAT;
#pragma unroll 1
  for (int kk = 0; kk < FEAT; kk += 32) {
    const int kb = kk + 8 * hi;                   // A-layout K base per lane half
    const float4 f0 = *(const float4*)(xr + kb);
    const float4 f1 = *(const float4*)(xr + kb + 4);
    const float4 f2 = *(const float4*)(xr + kb + 16);
    const float4 f3 = *(const float4*)(xr + kb + 20);
    v16bf a;
    a[0] = (__bf16)f0.x; a[1] = (__bf16)f0.y; a[2]  = (__bf16)f0.z; a[3]  = (__bf16)f0.w;
    a[4] = (__bf16)f1.x; a[5] = (__bf16)f1.y; a[6]  = (__bf16)f1.z; a[7]  = (__bf16)f1.w;
    a[8] = (__bf16)f2.x; a[9] = (__bf16)f2.y; a[10] = (__bf16)f2.z; a[11] = (__bf16)f2.w;
    a[12] = (__bf16)f3.x; a[13] = (__bf16)f3.y; a[14] = (__bf16)f3.z; a[15] = (__bf16)f3.w;
    v16bf b;                                       // B: lane = N col, elems = K (A order)
#pragma unroll
    for (int s = 0; s < 16; ++s)
      b[s] = W1b[(size_t)(kb + (s & 7) + ((s >> 3) << 4)) * NG + n0 + m16];
    c = __builtin_amdgcn_wmma_f32_16x16x32_bf16(false, a, false, b,
                                                (short)0, c, false, false);
  }

  // Scan-blocked store: tile rows = 16 consecutive t of one batch row b.
  const int b  = tM >> 6;                          // tM*16/1024
  const int t0 = (tM & 63) * 16;
  const int wg = b >> 4;
  const int mb = b & 15;
#pragma unroll
  for (int r = 0; r < 8; ++r) {
    const int t = t0 + r + 8 * hi;
    Z1[(size_t)(wg * TIME + t) * ZBLK + mb * NG + n0 + m16] = c[r];
  }
}

// ---------------------------------------------------------------------------
// Scan-kernel helpers
// ---------------------------------------------------------------------------
struct AFrag { v16bf a[4]; };                      // h @ [K=128 pad] A operands

__device__ __forceinline__ AFrag load_a(const __bf16* hs, int m16, int hi) {
  AFrag A;
#pragma unroll
  for (int ks = 0; ks < 4; ++ks) {
    const int kb = ks * 32 + 8 * hi;
    const v8bf lo = *(const v8bf*)(hs + m16 * HPAD + kb);
    const v8bf hi8 = *(const v8bf*)(hs + m16 * HPAD + kb + 16);
#pragma unroll
    for (int q = 0; q < 8; ++q) { A.a[ks][q] = lo[q]; A.a[ks][8 + q] = hi8[q]; }
  }
  return A;
}

__device__ __forceinline__ v8f
rec_gemm_acc(v8f c, const AFrag& A, const __bf16* __restrict__ W,
             int n0, int lane) {
#pragma unroll
  for (int ks = 0; ks < 3; ++ks) {                 // krow <= 95
    v16bf b;
#pragma unroll
    for (int s = 0; s < 16; ++s)
      b[s] = W[(size_t)(ks * 32 + 8 * (lane >> 4) + (s & 7) + ((s >> 3) << 4)) * NG + n0 + (lane & 15)];
    c = __builtin_amdgcn_wmma_f32_16x16x32_bf16(false, A.a[ks], false, b,
                                                (short)0, c, false, false);
  }
  {                                                // K tail: zero rows >= 100
    v16bf b;
#pragma unroll
    for (int q = 0; q < 16; ++q) {
      const int krow = 96 + 8 * (lane >> 4) + (q & 7) + ((q >> 3) << 4);
      b[q] = (krow < UNITS) ? W[(size_t)krow * NG + n0 + (lane & 15)] : (__bf16)0.0f;
    }
    c = __builtin_amdgcn_wmma_f32_16x16x32_bf16(false, A.a[3], false, b,
                                                (short)0, c, false, false);
  }
  return c;
}

// Async-prefetch one (wg,t) slice of Z1pre (25600 B) into LDS zin.
// 1600 x 16B chunks spread over 256 lanes; tracked by ASYNCcnt.
__device__ __forceinline__ void
async_fetch_zin(const float* __restrict__ src, unsigned zin_lds, int tid) {
  const uint64_t sbase = (uint64_t)(uintptr_t)src;
#pragma unroll
  for (int k = 0; k < 7; ++k) {
    const int chunk = tid + 256 * k;
    if (chunk < ZBLKB / 16) {
      const unsigned dst  = zin_lds + chunk * 16;
      const unsigned voff = chunk * 16;
      asm volatile("global_load_async_to_lds_b128 %0, %1, %2 offset:0"
                   :: "v"(dst), "v"(voff), "s"(sbase)
                   : "memory");
    }
  }
}

__device__ __forceinline__ void wait_async0() {
  asm volatile("s_wait_asynccnt 0" ::: "memory");
}

// ---------------------------------------------------------------------------
// Kernel 2: sequential scan. 8 workgroups x 256 threads (8 waves).
// Each WG owns 16 batch rows; h in LDS (bf16), c in regs, z staged in LDS,
// next step's Z1pre slice async-prefetched into LDS.
// ---------------------------------------------------------------------------
__global__ void __launch_bounds__(256)
lstm_scan(const float* __restrict__ Z1,
          const __bf16* __restrict__ U1b, const __bf16* __restrict__ W2b,
          const __bf16* __restrict__ U2b, const float* __restrict__ b2,
          float* __restrict__ out) {
  __shared__ __align__(32) __bf16 h1s[16 * HPAD];
  __shared__ __align__(32) __bf16 h2s[16 * HPAD];
  __shared__ __align__(32) float  zs[ZBLK];        // gate pre-activations
  __shared__ __align__(32) float  zin[ZBLK];       // prefetched Z1pre slice

  const int tid  = threadIdx.x;
  const int lane = tid & 31;
  const int wave = tid >> 5;
  const int m16  = lane & 15;
  const int hi   = lane >> 4;
  const int bb   = blockIdx.x * 16;                // batch base
  const unsigned zin_lds = (unsigned)(uintptr_t)&zin[0];
  const float* zsrc = Z1 + (size_t)blockIdx.x * TIME * ZBLK;

  for (int i = tid; i < 16 * HPAD; i += 256) {     // zero h (incl. K padding)
    h1s[i] = (__bf16)0.0f;
    h2s[i] = (__bf16)0.0f;
  }

  // Per-thread cell state + layer-2 biases, fixed (m,u) ownership.
  float c1r[7], c2r[7], bi[7], bff[7], bg[7], bo[7];
#pragma unroll
  for (int j = 0; j < 7; ++j) {
    c1r[j] = 0.0f; c2r[j] = 0.0f;
    bi[j] = bff[j] = bg[j] = bo[j] = 0.0f;
    const int e = tid + 256 * j;
    if (e < 16 * UNITS) {
      const int u = e % UNITS;
      bi[j]  = b2[u];
      bff[j] = b2[UNITS + u];
      bg[j]  = b2[2 * UNITS + u];
      bo[j]  = b2[3 * UNITS + u];
    }
  }

  async_fetch_zin(zsrc, zin_lds, tid);             // prefetch t = 0
  wait_async0();
  __syncthreads();

  for (int t = 0; t < TIME; ++t) {
    // ---------- layer 1 GEMM: z = zin + h1 @ U1 ----------
    {
      const AFrag A1 = load_a(h1s, m16, hi);
      for (int tile = wave; tile < NT; tile += 8) {
        const int n0 = tile * 16;
        v8f c;
#pragma unroll
        for (int r = 0; r < 8; ++r) c[r] = zin[(r + 8 * hi) * NG + n0 + m16];
        c = rec_gemm_acc(c, A1, U1b, n0, lane);
#pragma unroll
        for (int r = 0; r < 8; ++r) zs[(r + 8 * hi) * NG + n0 + m16] = c[r];
      }
    }
    __syncthreads();                               // zin consumed, zs ready

    // Overlap: prefetch next step's Z1pre slice while gates + layer 2 run.
    if (t + 1 < TIME)
      async_fetch_zin(zsrc + (size_t)(t + 1) * ZBLK, zin_lds, tid);

    // ---------- layer 1 gates ----------
#pragma unroll
    for (int j = 0; j < 7; ++j) {
      const int e = tid + 256 * j;
      if (e < 16 * UNITS) {
        const int m = e / UNITS, u = e - m * UNITS;
        const float zi = zs[m * NG + u];
        const float zf = zs[m * NG + UNITS + u];
        const float zg = zs[m * NG + 2 * UNITS + u];
        const float zo = zs[m * NG + 3 * UNITS + u];
        const float ig = fast_sigmoid(zi);
        const float fg = fast_sigmoid(zf);
        const float gg = fast_tanh(zg);
        const float og = fast_sigmoid(zo);
        const float cn = fg * c1r[j] + ig * gg;
        c1r[j] = cn;
        h1s[m * HPAD + u] = (__bf16)(og * fast_tanh(cn));
      }
    }
    __syncthreads();

    // ---------- layer 2 GEMM: z = h1 @ W2 + h2 @ U2 ----------
    {
      const AFrag Ah1 = load_a(h1s, m16, hi);
      const AFrag Ah2 = load_a(h2s, m16, hi);
      for (int tile = wave; tile < NT; tile += 8) {
        const int n0 = tile * 16;
        v8f c;
#pragma unroll
        for (int r = 0; r < 8; ++r) c[r] = 0.0f;
        c = rec_gemm_acc(c, Ah1, W2b, n0, lane);
        c = rec_gemm_acc(c, Ah2, U2b, n0, lane);
#pragma unroll
        for (int r = 0; r < 8; ++r) zs[(r + 8 * hi) * NG + n0 + m16] = c[r];
      }
    }
    __syncthreads();

    // ---------- layer 2 gates (+ b2) ----------
#pragma unroll
    for (int j = 0; j < 7; ++j) {
      const int e = tid + 256 * j;
      if (e < 16 * UNITS) {
        const int m = e / UNITS, u = e - m * UNITS;
        const float zi = zs[m * NG + u] + bi[j];
        const float zf = zs[m * NG + UNITS + u] + bff[j];
        const float zg = zs[m * NG + 2 * UNITS + u] + bg[j];
        const float zo = zs[m * NG + 3 * UNITS + u] + bo[j];
        const float ig = fast_sigmoid(zi);
        const float fg = fast_sigmoid(zf);
        const float gg = fast_tanh(zg);
        const float og = fast_sigmoid(zo);
        const float cn = fg * c2r[j] + ig * gg;
        c2r[j] = cn;
        const float hv = og * fast_tanh(cn);
        h2s[m * HPAD + u] = (__bf16)hv;
        if (t == TIME - 1) out[(bb + m) * UNITS + u] = hv;
      }
    }
    wait_async0();                                 // zin(t+1) landed
    __syncthreads();
  }
}

// ---------------------------------------------------------------------------
extern "C" void kernel_launch(void* const* d_in, const int* in_sizes, int n_in,
                              void* d_out, int out_size, void* d_ws, size_t ws_size,
                              hipStream_t stream) {
  (void)in_sizes; (void)n_in; (void)out_size; (void)ws_size;
  const float* x  = (const float*)d_in[0];
  const float* W1 = (const float*)d_in[1];
  const float* U1 = (const float*)d_in[2];
  const float* b1 = (const float*)d_in[3];
  const float* W2 = (const float*)d_in[4];
  const float* U2 = (const float*)d_in[5];
  const float* b2 = (const float*)d_in[6];
  float* out = (float*)d_out;

  char* ws = (char*)d_ws;
  const size_t zbytes = (size_t)BATCH * TIME * NG * sizeof(float); // 209.7 MB
  float*  Z1  = (float*)ws;
  __bf16* W1b = (__bf16*)(ws + zbytes);                 // 256*400*2 = 204800 B
  __bf16* U1b = (__bf16*)(ws + zbytes + 204800);
  __bf16* W2b = (__bf16*)(ws + zbytes + 204800 + 80128);
  __bf16* U2b = (__bf16*)(ws + zbytes + 204800 + 2 * 80128);

  convert_weights<<<400, 256, 0, stream>>>(W1, U1, W2, U2, W1b, U1b, W2b, U2b);

  // M tiles = 131072/16 = 8192, N tiles = 25 -> 204800 wave-tiles / 8 waves
  input_proj<<<(BATCH * TIME / 16) * NT / 8, 256, 0, stream>>>(x, W1b, b1, Z1);

  lstm_scan<<<BATCH / 16, 256, 0, stream>>>(Z1, U1b, W2b, U2b, b2, out);
}
